// MoEFFN_46883863003248
// MI455X (gfx1250) — matrix operand order repaired and checked by reading, not compiled, not run on hardware
//
#include <hip/hip_runtime.h>
#include <math.h>

// MoE FFN, top-2 gathered expert GEMMs on CDNA5 WMMA (bf16 in, f32 acc).
// B=4 T=2048 D=1024 E=8 H=4096

#define NTOK   8192
#define DDIM   1024
#define NEXP   8
#define HDIM   4096
#define TILE_M 128
#define CHU    128     // H-chunk width == X K-staging chunk width
#define UPITCH 136     // bf16 pitch (272B = 68 dwords ≡ 4 mod 64 -> no LDS bank conflicts)

typedef __bf16 bf16_t;
typedef __attribute__((ext_vector_type(16))) __bf16 v16bf;
typedef __attribute__((ext_vector_type(8)))  float  v8f;
typedef __attribute__((ext_vector_type(4)))  unsigned int u32x4;
typedef __attribute__((ext_vector_type(4)))  float  f32x4;

static __device__ inline v8f vzero8() {
  v8f z = {0.f, 0.f, 0.f, 0.f, 0.f, 0.f, 0.f, 0.f};
  return z;
}

// A fragment (16x32 bf16, row-major tile in LDS).
// Layout: lanes 0-15 hold K = k0+0..15 of row M=lane;
// lanes 16-31 hold K = k0+16..31. Two aligned 16B runs per lane.
static __device__ inline v16bf load_afrag(const bf16_t* base, int pitch, int m0, int k0, int lane) {
  const int m    = m0 + (lane & 15);
  const int half = lane >> 4;
  const bf16_t* p = base + m * pitch + k0 + half * 16;
  union { u32x4 u[2]; v16bf v; } t;
  t.u[0] = *(const u32x4*)(p);
  t.u[1] = *(const u32x4*)(p + 8);
  return t.v;
}

// B fragment (32x16) from f32 global, K-major with leading dim ld.
// Lanes 0-15: col n0+lane, K = k0+0..15; lanes 16-31: K = k0+16..31.
static __device__ inline v16bf load_bfrag(const float* base, int ld, int k0, int n0, int lane) {
  const int n  = n0 + (lane & 15);
  const int kb = k0 + ((lane >> 4) << 4);
  v16bf b;
  #pragma unroll
  for (int i = 0; i < 16; ++i)
    b[i] = (bf16_t)base[(long)(kb + i) * ld + n];
  return b;
}

// ---------------------------------------------------------------------------
__global__ void __launch_bounds__(256) init_kernel(float* out, int* hdr, int total) {
  int idx = blockIdx.x * blockDim.x + threadIdx.x;
  int stride = gridDim.x * blockDim.x;
  for (int i = idx; i < total; i += stride) out[i] = 0.f;
  if (idx < 32) hdr[idx] = 0;
}

// ---------------------------------------------------------------------------
// One wave (32 lanes) per token: logits over E=8, softmax, top-2, list build.
__global__ void __launch_bounds__(256) router_kernel(
    const float* __restrict__ x, const float* __restrict__ gW,
    int* cnt, int* loadcnt, float* imp,
    int* tokE, float* tokG, int* list, float* gwArr)
{
  const int lane = threadIdx.x & 31;
  const int wv   = threadIdx.x >> 5;
  const int tok  = blockIdx.x * 8 + wv;

  float acc[NEXP];
  #pragma unroll
  for (int e = 0; e < NEXP; ++e) acc[e] = 0.f;

  const float* xr = x + (long)tok * DDIM;
  for (int i = 0; i < DDIM / 32; ++i) {
    int d = lane + 32 * i;
    float xv = xr[d];
    const float* gr = gW + d * NEXP;
    #pragma unroll
    for (int e = 0; e < NEXP; ++e) acc[e] = fmaf(xv, gr[e], acc[e]);
  }
  #pragma unroll
  for (int off = 16; off > 0; off >>= 1) {
    #pragma unroll
    for (int e = 0; e < NEXP; ++e) acc[e] += __shfl_xor(acc[e], off, 32);
  }

  if (lane == 0) {
    float mx = acc[0];
    #pragma unroll
    for (int e = 1; e < NEXP; ++e) mx = fmaxf(mx, acc[e]);
    float p[NEXP]; float s = 0.f;
    #pragma unroll
    for (int e = 0; e < NEXP; ++e) { p[e] = expf(acc[e] - mx); s += p[e]; }
    float inv = 1.f / s;
    #pragma unroll
    for (int e = 0; e < NEXP; ++e) p[e] *= inv;

    int i1 = 0;
    #pragma unroll
    for (int e = 1; e < NEXP; ++e) if (p[e] > p[i1]) i1 = e;   // ties -> lowest idx (matches top_k)
    int i2 = (i1 == 0) ? 1 : 0;
    #pragma unroll
    for (int e = 0; e < NEXP; ++e) if (e != i1 && p[e] > p[i2]) i2 = e;

    float denom = p[i1] + p[i2];
    float g1 = p[i1] / denom, g2 = p[i2] / denom;

    #pragma unroll
    for (int e = 0; e < NEXP; ++e) atomicAdd(&imp[e], p[e]);
    atomicAdd(&loadcnt[i1], 1);

    tokE[tok] = i1; tokE[NTOK + tok] = i2;
    tokG[tok] = g1; tokG[NTOK + tok] = g2;

    int q1 = atomicAdd(&cnt[i1], 1);
    list[i1 * NTOK + q1] = tok; gwArr[i1 * NTOK + q1] = g1;
    int q2 = atomicAdd(&cnt[i2], 1);
    list[i2 * NTOK + q2] = tok; gwArr[i2 * NTOK + q2] = g2;
  }
}

// ---------------------------------------------------------------------------
__global__ void aux_kernel(const int* loadcnt, const float* imp, float* out) {
  float s = 0.f;
  for (int e = 0; e < NEXP; ++e)
    s += (imp[e] / (float)NTOK) * ((float)loadcnt[e] / (float)NTOK);
  out[(long)NTOK * DDIM] = (float)NEXP * s * 0.01f;
}

// ---------------------------------------------------------------------------
// Grouped GEMM over gathered tokens. Block = (tile of 128 tokens, expert).
// 8 waves. Per H-chunk (128 cols): GEMM1 over K=D with X staged into LDS
// in 128-wide K chunks (bf16), then bias+ReLU into the SAME LDS buffer,
// then GEMM2 over K=chunk into y via gate-weighted atomics.
__global__ void __launch_bounds__(256) ffn_kernel(
    const float* __restrict__ x,  const float* __restrict__ w1,
    const float* __restrict__ b1, const float* __restrict__ w2,
    float* __restrict__ y, const int* __restrict__ cnt,
    const int* __restrict__ list, const float* __restrict__ gwArr)
{
  __shared__ __align__(16) bf16_t U[TILE_M * UPITCH];   // X chunk / H tile (time-shared)
  __shared__ int   sTok[TILE_M];
  __shared__ float sG[TILE_M];

  const int e    = blockIdx.y;
  const int tile = blockIdx.x;
  const int n_e  = cnt[e];
  if (tile * TILE_M >= n_e) return;                 // uniform per block
  const int rem   = n_e - tile * TILE_M;
  const int valid = rem < TILE_M ? rem : TILE_M;

  const int tid   = threadIdx.x;
  const int lane  = tid & 31;
  const int wv    = tid >> 5;
  const int half8 = (lane >> 4) << 3;

  if (tid < TILE_M) {
    int idx = tile * TILE_M + tid;
    int tok = 0; float g = 0.f;
    if (idx < n_e) { tok = list[e * NTOK + idx]; g = gwArr[e * NTOK + idx]; }
    sTok[tid] = tok; sG[tid] = g;
  }
  __syncthreads();

  const float* w1e = w1 + (long)e * DDIM * HDIM;    // [D, H] K-major for GEMM1
  const float* w2e = w2 + (long)e * HDIM * DDIM;    // [H, D] K-major for GEMM2

  const int  srow  = tid >> 1;                      // 2 threads per row stage X
  const int  scol  = (tid & 1) * (CHU / 2);
  const long xbase = (long)sTok[srow] * DDIM + scol;

  for (int hc = 0; hc < HDIM / CHU; ++hc) {
    // ---------------- GEMM1: h = relu(Xg @ w1[:, chunk] + b1) ----------------
    const int nA = hc * CHU + wv * 16;              // this wave's 16 H-cols
    v8f acc[8];
    #pragma unroll
    for (int m = 0; m < 8; ++m) acc[m] = vzero8();

    #pragma unroll 1
    for (int kc = 0; kc < DDIM / CHU; ++kc) {
      __syncthreads();                              // prior readers of U done
      {
        const f32x4* xr = (const f32x4*)(x + xbase + kc * CHU);
        bf16_t* dst = U + srow * UPITCH + scol;
        #pragma unroll
        for (int i = 0; i < CHU / 8; ++i) {
          f32x4 f = xr[i];
          dst[4*i+0] = (bf16_t)f.x; dst[4*i+1] = (bf16_t)f.y;
          dst[4*i+2] = (bf16_t)f.z; dst[4*i+3] = (bf16_t)f.w;
        }
      }
      __syncthreads();
      #pragma unroll
      for (int k2 = 0; k2 < CHU; k2 += 32) {
        v16bf bf = load_bfrag(w1e, HDIM, kc * CHU + k2, nA, lane);
        #pragma unroll
        for (int mt = 0; mt < 8; ++mt) {
          v16bf af = load_afrag(U, UPITCH, mt * 16, k2, lane);
          acc[mt] = __builtin_amdgcn_wmma_f32_16x16x32_bf16(
              false, af, false, bf, (short)0, acc[mt], false, false);
        }
      }
    }
    __syncthreads();
    {   // bias + ReLU -> U reused as H tile [128 x CHU] bf16
      const int nl = wv * 16 + (lane & 15);
      const float bias = b1[(long)e * HDIM + hc * CHU + nl];
      #pragma unroll
      for (int mt = 0; mt < 8; ++mt) {
        #pragma unroll
        for (int v = 0; v < 8; ++v) {
          float t = acc[mt][v] + bias;
          t = t > 0.f ? t : 0.f;
          U[(mt * 16 + half8 + v) * UPITCH + nl] = (bf16_t)t;
        }
      }
    }
    __syncthreads();
    // ---------------- GEMM2: y += g * (h_chunk @ w2[chunk, :]) ---------------
    #pragma unroll 1
    for (int nt = 0; nt < 8; ++nt) {
      const int nB = wv * 128 + nt * 16;            // this wave's 16 D-cols
      v8f acc2[8];
      #pragma unroll
      for (int m = 0; m < 8; ++m) acc2[m] = vzero8();
      #pragma unroll
      for (int k2 = 0; k2 < CHU; k2 += 32) {
        v16bf bf = load_bfrag(w2e, DDIM, hc * CHU + k2, nB, lane);
        #pragma unroll
        for (int mt = 0; mt < 8; ++mt) {
          v16bf af = load_afrag(U, UPITCH, mt * 16, k2, lane);
          acc2[mt] = __builtin_amdgcn_wmma_f32_16x16x32_bf16(
              false, af, false, bf, (short)0, acc2[mt], false, false);
        }
      }
      const int nl = nB + (lane & 15);
      #pragma unroll
      for (int mt = 0; mt < 8; ++mt) {
        #pragma unroll
        for (int v = 0; v < 8; ++v) {
          int r = mt * 16 + half8 + v;
          if (r < valid)
            atomicAdd(y + (long)sTok[r] * DDIM + nl, acc2[mt][v] * sG[r]);
        }
      }
    }
  }
}

// ---------------------------------------------------------------------------
__global__ void __launch_bounds__(256) bias2_kernel(
    float* y, const float* b2, const int* tokE, const float* tokG)
{
  int idx = blockIdx.x * 256 + threadIdx.x;         // < NTOK*DDIM
  int t = idx >> 10;
  int d = idx & (DDIM - 1);
  float v = y[idx];
  v += tokG[t]        * b2[tokE[t]        * DDIM + d];
  v += tokG[NTOK + t] * b2[tokE[NTOK + t] * DDIM + d];
  y[idx] = v;
}

// ---------------------------------------------------------------------------
extern "C" void kernel_launch(void* const* d_in, const int* in_sizes, int n_in,
                              void* d_out, int out_size, void* d_ws, size_t ws_size,
                              hipStream_t stream) {
  const float* x  = (const float*)d_in[0];
  const float* gW = (const float*)d_in[1];
  const float* w1 = (const float*)d_in[2];
  const float* b1 = (const float*)d_in[3];
  const float* w2 = (const float*)d_in[4];
  const float* b2 = (const float*)d_in[5];
  float* out = (float*)d_out;

  // workspace layout (4B units): [cnt 8][loadcnt 8][imp 8][pad 8]
  //   [tokE 2*NTOK][tokG 2*NTOK][list E*NTOK][gw E*NTOK]  (~655 KB)
  int*   hdr     = (int*)d_ws;
  int*   cnt     = hdr;
  int*   loadcnt = hdr + 8;
  float* imp     = (float*)(hdr + 16);
  int*   tokE    = hdr + 32;
  float* tokG    = (float*)(hdr + 32 + 2 * NTOK);
  int*   list    = hdr + 32 + 4 * NTOK;
  float* gwArr   = (float*)(hdr + 32 + 4 * NTOK + NEXP * NTOK);

  init_kernel<<<4096, 256, 0, stream>>>(out, hdr, NTOK * DDIM + 1);
  router_kernel<<<NTOK / 8, 256, 0, stream>>>(x, gW, cnt, loadcnt, imp,
                                              tokE, tokG, list, gwArr);
  aux_kernel<<<1, 1, 0, stream>>>(loadcnt, imp, out);
  dim3 g(NTOK / TILE_M, NEXP);
  ffn_kernel<<<g, 256, 0, stream>>>(x, w1, b1, w2, out, cnt, list, gwArr);
  bias2_kernel<<<(NTOK * DDIM) / 256, 256, 0, stream>>>(out, b2, tokE, tokG);
}
